// GraphCare_45183055954287
// MI455X (gfx1250) — matrix operand order repaired and checked by reading, not compiled, not run on hardware
//
#include <hip/hip_runtime.h>

#define B_   32
#define V_   20
#define NN_  4000
#define NR_  400
#define EMB_ 128
#define HID_ 128
#define L_   3
#define N_   32000
#define E_   256000
#define OUT_ 128

typedef __attribute__((ext_vector_type(16))) _Float16 v16h;
typedef __attribute__((ext_vector_type(8)))  _Float16 v8h;
typedef __attribute__((ext_vector_type(8)))  float    v8f;

__device__ __forceinline__ v8f zero8() {
  v8f r = {0.f, 0.f, 0.f, 0.f, 0.f, 0.f, 0.f, 0.f};
  return r;
}

__device__ __forceinline__ void atomAddF(float* p, float v) {
  unsafeAtomicAdd(p, v);   // native global_atomic_add_f32 (no return)
}

// ---------------------------------------------------------------------------
// WMMA fragment loads from pre-converted f16 buffers (per ISA VGPR layouts)
// A 16x32: lanes 0-15 row=sub K 0..7 + 16..23; lanes 16-31 row=sub K 8..15 + 24..31
// B 32x16: lane sub = column; lanes 0-15 K 0..7 + 16..23; lanes 16-31 K 8..15 + 24..31
// ---------------------------------------------------------------------------
__device__ __forceinline__ v16h loadA16(const _Float16* __restrict__ rowp, int k0, int half) {
  v8h lo = *(const v8h*)(rowp + k0 + half * 8);
  v8h hi = *(const v8h*)(rowp + k0 + 16 + half * 8);
  return __builtin_shufflevector(lo, hi, 0, 1, 2, 3, 4, 5, 6, 7,
                                 8, 9, 10, 11, 12, 13, 14, 15);
}
__device__ __forceinline__ v16h loadB16(const _Float16* __restrict__ rowp, int k0, int half) {
  v8h lo = *(const v8h*)(rowp + k0 + half * 8);
  v8h hi = *(const v8h*)(rowp + k0 + 16 + half * 8);
  return __builtin_shufflevector(lo, hi, 0, 1, 2, 3, 4, 5, 6, 7,
                                 8, 9, 10, 11, 12, 13, 14, 15);
}

// f32 -> f16 fragment helpers for the small K=128 update GEMM
__device__ __forceinline__ v16h frag_a_sum(const float* __restrict__ r0,
                                           const float* __restrict__ r1,
                                           int k0, int half) {
  const float4* a0 = (const float4*)(r0 + k0 + half * 8);
  const float4* a1 = (const float4*)(r0 + k0 + 16 + half * 8);
  const float4* b0 = (const float4*)(r1 + k0 + half * 8);
  const float4* b1 = (const float4*)(r1 + k0 + 16 + half * 8);
  float4 f0 = a0[0], f1 = a0[1], f2 = a1[0], f3 = a1[1];
  float4 g0 = b0[0], g1 = b0[1], g2 = b1[0], g3 = b1[1];
  v16h a;
  a[0]=(_Float16)(f0.x+g0.x); a[1]=(_Float16)(f0.y+g0.y); a[2]=(_Float16)(f0.z+g0.z); a[3]=(_Float16)(f0.w+g0.w);
  a[4]=(_Float16)(f1.x+g1.x); a[5]=(_Float16)(f1.y+g1.y); a[6]=(_Float16)(f1.z+g1.z); a[7]=(_Float16)(f1.w+g1.w);
  a[8]=(_Float16)(f2.x+g2.x); a[9]=(_Float16)(f2.y+g2.y); a[10]=(_Float16)(f2.z+g2.z); a[11]=(_Float16)(f2.w+g2.w);
  a[12]=(_Float16)(f3.x+g3.x); a[13]=(_Float16)(f3.y+g3.y); a[14]=(_Float16)(f3.z+g3.z); a[15]=(_Float16)(f3.w+g3.w);
  return a;
}
__device__ __forceinline__ v16h frag_b_f32(const float* __restrict__ colp, int k0, int half) {
  const float4* p = (const float4*)(colp + k0 + half * 8);
  const float4* q = (const float4*)(colp + k0 + 16 + half * 8);
  float4 f0 = p[0], f1 = p[1], f2 = q[0], f3 = q[1];
  v16h b;
  b[0]=(_Float16)f0.x; b[1]=(_Float16)f0.y; b[2]=(_Float16)f0.z; b[3]=(_Float16)f0.w;
  b[4]=(_Float16)f1.x; b[5]=(_Float16)f1.y; b[6]=(_Float16)f1.z; b[7]=(_Float16)f1.w;
  b[8]=(_Float16)f2.x; b[9]=(_Float16)f2.y; b[10]=(_Float16)f2.z; b[11]=(_Float16)f2.w;
  b[12]=(_Float16)f3.x; b[13]=(_Float16)f3.y; b[14]=(_Float16)f3.z; b[15]=(_Float16)f3.w;
  return b;
}

// ---------------------------------------------------------------------------
// f32 -> f16 bulk conversion (8 elements/thread, vectorized)
// ---------------------------------------------------------------------------
__global__ void k_cvt_h(const float* __restrict__ s, _Float16* __restrict__ d, int n8) {
  int gid = blockIdx.x * blockDim.x + threadIdx.x;
  if (gid >= n8) return;
  const float4* sp = (const float4*)s + (size_t)gid * 2;
  float4 f0 = sp[0], f1 = sp[1];
  v8h h;
  h[0]=(_Float16)f0.x; h[1]=(_Float16)f0.y; h[2]=(_Float16)f0.z; h[3]=(_Float16)f0.w;
  h[4]=(_Float16)f1.x; h[5]=(_Float16)f1.y; h[6]=(_Float16)f1.z; h[7]=(_Float16)f1.w;
  ((v8h*)d)[gid] = h;
}

// ---------------------------------------------------------------------------
// node_lin = node_emb @ lin_w.T + lin_b ; rel_lin = rel_emb @ lin_w.T + lin_b
// ---------------------------------------------------------------------------
__global__ void k_prep_lin(const float* __restrict__ node_emb,
                           const float* __restrict__ rel_emb,
                           const float* __restrict__ lin_w,
                           const float* __restrict__ lin_b,
                           float* __restrict__ node_lin,
                           float* __restrict__ rel_lin) {
  int gid = blockIdx.x * blockDim.x + threadIdx.x;
  if (gid >= (NN_ + NR_) * HID_) return;
  int row = gid >> 7, h = gid & 127;
  const float* e; float* o;
  if (row < NN_) { e = node_emb + (size_t)row * EMB_; o = node_lin + (size_t)row * HID_ + h; }
  else { int r = row - NN_; e = rel_emb + (size_t)r * EMB_; o = rel_lin + (size_t)r * HID_ + h; }
  const float* w = lin_w + (size_t)h * EMB_;
  float acc = lin_b[h];
  #pragma unroll 4
  for (int k = 0; k < EMB_; ++k) acc += e[k] * w[k];
  *o = acc;
}

// x[i] = node_lin[node_ids[i]]
__global__ void k_gather_x(const float* __restrict__ node_lin,
                           const int* __restrict__ node_ids,
                           float* __restrict__ x) {
  int gid = blockIdx.x * blockDim.x + threadIdx.x;
  int i = gid >> 5, lane = gid & 31;
  if (i >= N_) return;
  ((float4*)(x + (size_t)i * HID_))[lane] =
      ((const float4*)(node_lin + (size_t)node_ids[i] * HID_))[lane];
}

// beta[b,v] = tanh(visit[b,v,:]·beta_w + beta_b) * exp(0.03*(V-v))
__global__ void k_beta(const float* __restrict__ visit,
                       const float* __restrict__ bw,
                       const float* __restrict__ bb,
                       float* __restrict__ beta) {
  int bv = blockIdx.x;
  int v = bv % V_;
  const float* row = visit + (size_t)bv * NN_;
  float acc = 0.f;
  for (int n = threadIdx.x; n < NN_; n += 32) acc += row[n] * bw[n];
  for (int off = 16; off > 0; off >>= 1) acc += __shfl_down(acc, off, 32);
  if (threadIdx.x == 0)
    beta[bv] = tanhf(acc + bb[0]) * __expf(0.03f * (float)(V_ - v));
}

// wrel[r] = rel_lin[r]·wr_w + wr_b   (per relation)
__global__ void k_wrel(const float* __restrict__ rel_lin,
                       const float* __restrict__ ww,
                       const float* __restrict__ wb,
                       float* __restrict__ wrel) {
  int r = blockIdx.x * blockDim.x + threadIdx.x;
  if (r >= NR_) return;
  float acc = wb[0];
  const float* e = rel_lin + (size_t)r * HID_;
  #pragma unroll 4
  for (int k = 0; k < HID_; ++k) acc += e[k] * ww[k];
  wrel[r] = acc;
}

// ---------------------------------------------------------------------------
// z = visit(640x4000) @ alpha_w[l].T + alpha_b[l]   (f16 inputs, f32 accum)
// Wave computes a 32x32 output block (2x2 WMMA tiles): 4 wmma per K-step,
// 128 B/lane of f16 loads per 64 KFLOP. 2500 waves, grid 625 x 128.
// ---------------------------------------------------------------------------
__global__ __launch_bounds__(128) void k_gemm_z(const _Float16* __restrict__ Ah,
                                                const _Float16* __restrict__ Wh,
                                                const float* __restrict__ ab,
                                                float* __restrict__ z) {
  int wid  = blockIdx.x * 4 + (threadIdx.x >> 5);   // 0..2499
  int lane = threadIdx.x & 31;
  int half = lane >> 4, sub = lane & 15;
  int mg = wid / 125, ng = wid % 125;               // 20 x 125 groups of 32x32
  const _Float16* A0 = Ah + (size_t)(mg * 32 + sub) * NN_;
  const _Float16* A1 = A0 + (size_t)16 * NN_;
  const _Float16* W0 = Wh + (size_t)(ng * 32 + sub) * NN_;
  const _Float16* W1 = W0 + (size_t)16 * NN_;
  v8f c00 = zero8(), c01 = zero8(), c10 = zero8(), c11 = zero8();
  for (int k0 = 0; k0 < NN_; k0 += 32) {
    v16h a0 = loadA16(A0, k0, half);
    v16h a1 = loadA16(A1, k0, half);
    v16h b0 = loadB16(W0, k0, half);
    v16h b1 = loadB16(W1, k0, half);
    c00 = __builtin_amdgcn_wmma_f32_16x16x32_f16(false, a0, false, b0, (short)0, c00, false, false);
    c01 = __builtin_amdgcn_wmma_f32_16x16x32_f16(false, a0, false, b1, (short)0, c01, false, false);
    c10 = __builtin_amdgcn_wmma_f32_16x16x32_f16(false, a1, false, b0, (short)0, c10, false, false);
    c11 = __builtin_amdgcn_wmma_f32_16x16x32_f16(false, a1, false, b1, (short)0, c11, false, false);
  }
  int col0 = ng * 32 + sub, col1 = col0 + 16;
  float bi0 = ab[col0], bi1 = ab[col1];
  int r0 = mg * 32 + half * 8, r1 = r0 + 16;
  #pragma unroll
  for (int i = 0; i < 8; ++i) {
    z[(size_t)(r0 + i) * NN_ + col0] = c00[i] + bi0;
    z[(size_t)(r0 + i) * NN_ + col1] = c01[i] + bi1;
    z[(size_t)(r1 + i) * NN_ + col0] = c10[i] + bi0;
    z[(size_t)(r1 + i) * NN_ + col1] = c11[i] + bi1;
  }
}

// attn[b,m] = sum_v softmax_v(z[b,v,m]) * beta[b,v]
__global__ void k_attn(const float* __restrict__ z,
                       const float* __restrict__ beta,
                       float* __restrict__ attn) {
  int gid = blockIdx.x * blockDim.x + threadIdx.x;
  if (gid >= B_ * NN_) return;
  int b = gid / NN_, m = gid % NN_;
  const float* zp = z + (size_t)b * V_ * NN_ + m;
  float zv[V_];
  float mx = -3.4e38f;
  #pragma unroll
  for (int v = 0; v < V_; ++v) { zv[v] = zp[(size_t)v * NN_]; mx = fmaxf(mx, zv[v]); }
  float se = 0.f, acc = 0.f;
  #pragma unroll
  for (int v = 0; v < V_; ++v) {
    float e = __expf(zv[v] - mx);
    se += e;
    acc += e * beta[b * V_ + v];
  }
  attn[gid] = acc / se;
}

// msg = relu(x[src]*attn + wrel*rel_lin); scatter-add into agg[dst]
__global__ __launch_bounds__(256) void k_msg(const int* __restrict__ ei,
                                             const int* __restrict__ node_ids,
                                             const int* __restrict__ batch,
                                             const int* __restrict__ rel_ids,
                                             const float* __restrict__ attn,
                                             const float* __restrict__ wrel,
                                             const float* __restrict__ rel_lin,
                                             const float* __restrict__ x,
                                             float* __restrict__ agg) {
  int e    = blockIdx.x * 8 + (threadIdx.x >> 5);
  int lane = threadIdx.x & 31;
  int s = ei[e], d = ei[E_ + e];
  float ea = attn[batch[s] * NN_ + node_ids[s]];
  int r = rel_ids[e];
  float wr = wrel[r];
  float4 xv = ((const float4*)(x + (size_t)s * HID_))[lane];
  float4 rv = ((const float4*)(rel_lin + (size_t)r * HID_))[lane];
  float4 m;
  m.x = fmaxf(xv.x * ea + wr * rv.x, 0.f);
  m.y = fmaxf(xv.y * ea + wr * rv.y, 0.f);
  m.z = fmaxf(xv.z * ea + wr * rv.z, 0.f);
  m.w = fmaxf(xv.w * ea + wr * rv.w, 0.f);
  float* o = agg + (size_t)d * HID_ + lane * 4;
  atomAddF(o + 0, m.x); atomAddF(o + 1, m.y);
  atomAddF(o + 2, m.z); atomAddF(o + 3, m.w);
}

// x_out = relu((agg + x) @ nn_w[l].T + nn_b[l]); WMMA K=128
__global__ __launch_bounds__(256) void k_update(const float* __restrict__ agg,
                                                const float* __restrict__ x,
                                                const float* __restrict__ nnw,
                                                const float* __restrict__ nnb,
                                                float* __restrict__ xo) {
  int mt   = blockIdx.x;
  int nt   = threadIdx.x >> 5;
  int lane = threadIdx.x & 31;
  int half = lane >> 4, sub = lane & 15;
  int row = mt * 16 + sub;
  const float* ar = agg + (size_t)row * HID_;
  const float* xr = x + (size_t)row * HID_;
  const float* wr = nnw + (size_t)(nt * 16 + sub) * HID_;
  v8f acc = zero8();
  #pragma unroll
  for (int k0 = 0; k0 < HID_; k0 += 32) {
    v16h a = frag_a_sum(ar, xr, k0, half);
    v16h b = frag_b_f32(wr, k0, half);
    acc = __builtin_amdgcn_wmma_f32_16x16x32_f16(false, a, false, b, (short)0, acc, false, false);
  }
  int h = nt * 16 + sub;
  float bi = nnb[h];
  int rbase = mt * 16 + half * 8;
  #pragma unroll
  for (int i = 0; i < 8; ++i)
    xo[(size_t)(rbase + i) * HID_ + h] = fmaxf(acc[i] + bi, 0.f);
}

// segment-sum pooling
__global__ void k_pool(const float* __restrict__ x,
                       const int* __restrict__ batch,
                       float* __restrict__ xg,
                       float* __restrict__ cnt) {
  int n    = blockIdx.x * 8 + (threadIdx.x >> 5);
  int lane = threadIdx.x & 31;
  int b = batch[n];
  float4 xv = ((const float4*)(x + (size_t)n * HID_))[lane];
  float* o = xg + b * HID_ + lane * 4;
  atomAddF(o + 0, xv.x); atomAddF(o + 1, xv.y);
  atomAddF(o + 2, xv.z); atomAddF(o + 3, xv.w);
  if (lane == 0) atomAddF(cnt + b, 1.0f);
}

// tmp[b,:] = (ehr[b] @ node_emb) / sum(ehr[b])
__global__ void k_xnode1(const float* __restrict__ ehr,
                         const float* __restrict__ node_emb,
                         float* __restrict__ tmp) {
  int b = blockIdx.x, e = threadIdx.x;
  const float* er = ehr + (size_t)b * NN_;
  float acc = 0.f, den = 0.f;
  for (int n = 0; n < NN_; ++n) {
    float w = er[n];
    den += w;
    acc += w * node_emb[(size_t)n * EMB_ + e];
  }
  tmp[b * EMB_ + e] = acc / den;
}

// x_node = tmp @ lin_w.T + lin_b
__global__ void k_xnode2(const float* __restrict__ tmp,
                         const float* __restrict__ lin_w,
                         const float* __restrict__ lin_b,
                         float* __restrict__ xn) {
  int gid = blockIdx.x * blockDim.x + threadIdx.x;
  if (gid >= B_ * HID_) return;
  int b = gid >> 7, h = gid & 127;
  float acc = lin_b[h];
  const float* t = tmp + b * EMB_;
  const float* w = lin_w + (size_t)h * EMB_;
  #pragma unroll 4
  for (int k = 0; k < EMB_; ++k) acc += t[k] * w[k];
  xn[gid] = acc;
}

// logits = concat(xg/cnt, x_node) @ mlp_w.T + mlp_b
__global__ void k_final(const float* __restrict__ xg,
                        const float* __restrict__ cnt,
                        const float* __restrict__ xn,
                        const float* __restrict__ mw,
                        const float* __restrict__ mb,
                        float* __restrict__ out) {
  int gid = blockIdx.x * blockDim.x + threadIdx.x;
  if (gid >= B_ * OUT_) return;
  int b = gid >> 7, o = gid & 127;
  float inv = 1.0f / fmaxf(cnt[b], 1.0f);
  const float* w = mw + (size_t)o * (2 * HID_);
  float acc = mb[o];
  #pragma unroll 4
  for (int h = 0; h < HID_; ++h) acc += (xg[b * HID_ + h] * inv) * w[h];
  #pragma unroll 4
  for (int h = 0; h < HID_; ++h) acc += xn[b * HID_ + h] * w[HID_ + h];
  out[b * OUT_ + o] = acc;
}

// ---------------------------------------------------------------------------
static inline char* carveB(char*& p, size_t bytes) {
  char* r = p;
  p += ((bytes + 255) / 256) * 256;
  return r;
}
static inline float* carveF(char*& p, size_t nfloats) {
  return (float*)carveB(p, nfloats * sizeof(float));
}

extern "C" void kernel_launch(void* const* d_in, const int* in_sizes, int n_in,
                              void* d_out, int out_size, void* d_ws, size_t ws_size,
                              hipStream_t stream) {
  const int*   node_ids = (const int*)d_in[0];
  const int*   rel_ids  = (const int*)d_in[1];
  const int*   edge_idx = (const int*)d_in[2];
  const int*   batch    = (const int*)d_in[3];
  const float* visit    = (const float*)d_in[4];
  const float* ehr      = (const float*)d_in[5];
  const float* node_emb = (const float*)d_in[6];
  const float* rel_emb  = (const float*)d_in[7];
  const float* lin_w    = (const float*)d_in[8];
  const float* lin_b    = (const float*)d_in[9];
  const float* alpha_w  = (const float*)d_in[10];
  const float* alpha_b  = (const float*)d_in[11];
  const float* beta_w   = (const float*)d_in[12];
  const float* beta_b   = (const float*)d_in[13];
  const float* wr_w     = (const float*)d_in[14];
  const float* wr_b     = (const float*)d_in[15];
  const float* nn_w     = (const float*)d_in[16];
  const float* nn_b     = (const float*)d_in[17];
  const float* mlp_w    = (const float*)d_in[18];
  const float* mlp_b    = (const float*)d_in[19];
  float* out = (float*)d_out;

  char* p = (char*)d_ws;
  float*     rel_lin  = carveF(p, (size_t)NR_ * HID_);
  float*     node_lin = carveF(p, (size_t)NN_ * HID_);
  float*     xA       = carveF(p, (size_t)N_ * HID_);
  float*     xB       = carveF(p, (size_t)N_ * HID_);
  float*     agg      = carveF(p, (size_t)N_ * HID_);
  float*     z        = carveF(p, (size_t)B_ * V_ * NN_);
  float*     attn     = carveF(p, (size_t)B_ * NN_);
  float*     beta     = carveF(p, (size_t)B_ * V_);
  float*     wrel     = carveF(p, (size_t)NR_);
  float*     xg       = carveF(p, (size_t)B_ * HID_);
  float*     cnts     = carveF(p, (size_t)B_);
  float*     tmpn     = carveF(p, (size_t)B_ * EMB_);
  float*     xnode    = carveF(p, (size_t)B_ * HID_);
  _Float16*  Ah16     = (_Float16*)carveB(p, (size_t)B_ * V_ * NN_ * 2);
  _Float16*  Wh16     = (_Float16*)carveB(p, (size_t)NN_ * NN_ * 2);

  // 1. project embeddings once, then gather; convert visit to f16 once
  k_prep_lin<<<((NN_ + NR_) * HID_ + 255) / 256, 256, 0, stream>>>(
      node_emb, rel_emb, lin_w, lin_b, node_lin, rel_lin);
  k_gather_x<<<(N_ * 32) / 256, 256, 0, stream>>>(node_lin, node_ids, xA);
  {
    int n8 = (B_ * V_ * NN_) / 8;
    k_cvt_h<<<(n8 + 255) / 256, 256, 0, stream>>>(visit, Ah16, n8);
  }

  // 2. L GNN layers
  for (int l = 0; l < L_; ++l) {
    float* xc = (l & 1) ? xB : xA;
    float* xn = (l & 1) ? xA : xB;
    k_beta<<<B_ * V_, 32, 0, stream>>>(visit, beta_w + (size_t)l * NN_, beta_b + l, beta);
    k_wrel<<<(NR_ + 255) / 256, 256, 0, stream>>>(rel_lin, wr_w + (size_t)l * HID_,
                                                  wr_b + l, wrel);
    {
      int n8 = (NN_ * NN_) / 8;
      k_cvt_h<<<(n8 + 255) / 256, 256, 0, stream>>>(alpha_w + (size_t)l * NN_ * NN_,
                                                    Wh16, n8);
    }
    k_gemm_z<<<(20 * 125) / 4, 128, 0, stream>>>(Ah16, Wh16,
                                                 alpha_b + (size_t)l * NN_, z);
    k_attn<<<(B_ * NN_ + 255) / 256, 256, 0, stream>>>(z, beta, attn);
    hipMemsetAsync(agg, 0, (size_t)N_ * HID_ * sizeof(float), stream);
    k_msg<<<E_ / 8, 256, 0, stream>>>(edge_idx, node_ids, batch, rel_ids,
                                      attn, wrel, rel_lin, xc, agg);
    k_update<<<N_ / 16, 256, 0, stream>>>(agg, xc, nn_w + (size_t)l * HID_ * HID_,
                                          nn_b + (size_t)l * HID_, xn);
  }
  float* xfin = xB;  // l=0: A->B, l=1: B->A, l=2: A->B

  // 3. pooling + node-level embedding + final MLP
  hipMemsetAsync(xg, 0, (size_t)B_ * HID_ * sizeof(float), stream);
  hipMemsetAsync(cnts, 0, (size_t)B_ * sizeof(float), stream);
  k_pool<<<N_ / 8, 256, 0, stream>>>(xfin, batch, xg, cnts);
  k_xnode1<<<B_, EMB_, 0, stream>>>(ehr, node_emb, tmpn);
  k_xnode2<<<(B_ * HID_ + 255) / 256, 256, 0, stream>>>(tmpn, lin_w, lin_b, xnode);
  k_final<<<(B_ * OUT_ + 255) / 256, 256, 0, stream>>>(xg, cnts, xnode, mlp_w, mlp_b, out);

  (void)in_sizes; (void)n_in; (void)out_size; (void)ws_size;
}
